// CannyEdgeDetector_9285719294464
// MI455X (gfx1250) — matrix operand order repaired and checked by reading, not compiled, not run on hardware
//
#include <hip/hip_runtime.h>
#include <hip/hip_bf16.h>

typedef __attribute__((ext_vector_type(16))) _Float16 v16h;
typedef __attribute__((ext_vector_type(8)))  float    v8f;

#define NB   32
#define NH   512
#define NW   512
#define NPIX (NB * NH * NW)

// numpy-style 'reflect' (no edge repeat): -1 -> 1, N -> N-2
__device__ __forceinline__ int refl(int i, int n) {
    if (i < 0)  i = -i;
    if (i >= n) i = 2 * n - 2 - i;
    return i;
}

// Gaussian 5-tap, sigma = 1.4, normalized (f64-derived constants)
__device__ __forceinline__ float gw5(int d) {
    // d in [0,4]
    return (d == 2) ? 0.3057569f : ((d == 1 || d == 3) ? 0.2369121f : 0.1102096f);
}

// ---------------------------------------------------------------------------
// Fused grayscale + separable 5x5 Gaussian blur via WMMA banded matmuls.
// One wave per 16x16 output tile. grid = (32 tiles_x, 32 tiles_y, 32 batch).
// ---------------------------------------------------------------------------
__global__ __launch_bounds__(32) void k_blur(const float* __restrict__ x,
                                             float* __restrict__ blur)
{
    const int lane = threadIdx.x;      // 0..31 (wave32)
    const int half = lane >> 4;        // 0 or 1
    const int l16  = lane & 15;
    const int tx = blockIdx.x, ty = blockIdx.y, b = blockIdx.z;

    // Horizontal band matrix B (32x16 f16): B[k][j] = g[k-j], k-j in [0,5)
    // B fragment: col N = lane&15, element i -> K = (i<8 ? i : i+8) + half*8
    v16h bh;
#pragma unroll
    for (int i = 0; i < 16; ++i) {
        int k = (i < 8 ? i : i + 8) + half * 8;
        int d = k - l16;
        bh[i] = (d >= 0 && d < 5) ? (_Float16)gw5(d) : (_Float16)0.0f;
    }

    __shared__ float hb[32][17];   // h-blurred strip (rows ty*16-2 .. +29), padded

    const float* xr = x + (size_t)b * 3 * NH * NW;
    const int colbase = tx * 16 - 2;

    // Two horizontal WMMAs covering 32 rows of h-blur (need rows -2..+17)
#pragma unroll
    for (int s = 0; s < 2; ++s) {
        const int grow = refl(ty * 16 - 2 + s * 16 + l16, NH);
        const float* r0 = xr + (size_t)grow * NW;
        // A fragment: row M = lane&15, element i -> K per ISA layout
        v16h a;
#pragma unroll
        for (int i = 0; i < 16; ++i) {
            int kk = (i < 8 ? i : i + 8) + half * 8;      // K index 0..31
            int gc = refl(colbase + kk, NW);
            float r = r0[gc];
            float g = r0[(size_t)NH * NW + gc];
            float bb = r0[(size_t)2 * NH * NW + gc];
            a[i] = (_Float16)(0.299f * r + 0.587f * g + 0.114f * bb);
        }
        v8f cz = {};
        v8f c = __builtin_amdgcn_wmma_f32_16x16x32_f16(
            false, a, false, bh, (short)0, cz, false, false);
        // C layout: VGPR v -> M = v + half*8, N = lane&15
#pragma unroll
        for (int v = 0; v < 8; ++v)
            hb[s * 16 + v + half * 8][l16] = c[v];
    }
    __syncthreads();

    // Vertical band matrix A (16x32 f16): A[m][k] = g[k-m]
    v16h av;
#pragma unroll
    for (int i = 0; i < 16; ++i) {
        int kk = (i < 8 ? i : i + 8) + half * 8;
        int d = kk - l16;                       // row M = lane&15
        av[i] = (d >= 0 && d < 5) ? (_Float16)gw5(d) : (_Float16)0.0f;
    }
    // B fragment from staged h-blur: col N = lane&15, element i -> K = (i<8 ? i : i+8) + half*8
    v16h bv;
#pragma unroll
    for (int i = 0; i < 16; ++i) {
        int k = (i < 8 ? i : i + 8) + half * 8;
        bv[i] = (_Float16)hb[k][l16];
    }
    v8f cz2 = {};
    v8f c2 = __builtin_amdgcn_wmma_f32_16x16x32_f16(
        false, av, false, bv, (short)0, cz2, false, false);

    float* ob = blur + (size_t)b * NH * NW;
#pragma unroll
    for (int v = 0; v < 8; ++v) {
        int gy = ty * 16 + v + half * 8;
        ob[(size_t)gy * NW + tx * 16 + l16] = c2[v];
    }
}

// ---------------------------------------------------------------------------
// Sobel + magnitude + angle class + per-image max (atomic on uint bits)
// ---------------------------------------------------------------------------
__global__ void k_init_max(unsigned int* mmax)
{
    if (threadIdx.x < NB) mmax[threadIdx.x] = 0u;
}

__global__ void k_sobel(const float* __restrict__ blur, float* __restrict__ mag,
                        unsigned char* __restrict__ cls, unsigned int* __restrict__ mmax)
{
    const int i = blockIdx.x * blockDim.x + threadIdx.x;
    if (i >= NPIX) return;
    const int b = i / (NH * NW);
    const int rem = i - b * (NH * NW);
    const int y = rem / NW, x = rem - y * NW;
    const float* p = blur + (size_t)b * NH * NW;
    const int ym = refl(y - 1, NH), yp = refl(y + 1, NH);
    const int xm = refl(x - 1, NW), xp = refl(x + 1, NW);

    float a00 = p[(size_t)ym * NW + xm], a01 = p[(size_t)ym * NW + x], a02 = p[(size_t)ym * NW + xp];
    float a10 = p[(size_t)y  * NW + xm],                               a12 = p[(size_t)y  * NW + xp];
    float a20 = p[(size_t)yp * NW + xm], a21 = p[(size_t)yp * NW + x], a22 = p[(size_t)yp * NW + xp];

    float gx = (a02 + 2.f * a12 + a22) - (a00 + 2.f * a10 + a20);
    float gy = (a20 + 2.f * a21 + a22) - (a00 + 2.f * a01 + a02);
    float m  = hypotf(gx, gy);
    mag[i] = m;

    float ang = fmodf(atan2f(gy, gx) * 57.29577951308232f + 180.0f, 180.0f);
    unsigned char c = 0;
    if      (ang >= 22.5f  && ang < 67.5f)  c = 1;
    else if (ang >= 67.5f  && ang < 112.5f) c = 2;
    else if (ang >= 112.5f && ang < 157.5f) c = 3;
    cls[i] = c;

    // wave32 max reduction, then one atomic per wave (mag >= 0 so uint order == float order)
    float wm = m;
#pragma unroll
    for (int off = 16; off; off >>= 1)
        wm = fmaxf(wm, __shfl_xor(wm, off, 32));
    if ((threadIdx.x & 31) == 0)
        atomicMax(&mmax[b], __float_as_uint(wm));
}

// ---------------------------------------------------------------------------
// NMS + double threshold (zero padding for neighbors, interior mask)
// ---------------------------------------------------------------------------
__global__ void k_nms(const float* __restrict__ mag, const unsigned char* __restrict__ cls,
                      const unsigned int* __restrict__ mmax,
                      unsigned char* __restrict__ strong, unsigned char* __restrict__ weak)
{
    const int i = blockIdx.x * blockDim.x + threadIdx.x;
    if (i >= NPIX) return;
    const int b = i / (NH * NW);
    const int rem = i - b * (NH * NW);
    const int y = rem / NW, x = rem - y * NW;
    const float* p = mag + (size_t)b * NH * NW;

    auto L = [&](int yy, int xx) -> float {
        return (yy < 0 || yy >= NH || xx < 0 || xx >= NW) ? 0.0f : p[(size_t)yy * NW + xx];
    };

    const float m = p[(size_t)y * NW + x];
    const unsigned char c = cls[i];
    float q, r;
    if      (c == 1) { q = L(y + 1, x - 1); r = L(y - 1, x + 1); }   // SW / NE
    else if (c == 2) { q = L(y + 1, x);     r = L(y - 1, x);     }   // S  / N
    else if (c == 3) { q = L(y - 1, x - 1); r = L(y + 1, x + 1); }   // NW / SE
    else             { q = L(y,     x + 1); r = L(y,     x - 1); }   // E  / W

    const bool keep = (m >= q) && (m >= r) &&
                      (y >= 1 && y <= NH - 2 && x >= 1 && x <= NW - 2);
    const float mx = __uint_as_float(mmax[b]);
    const float mn = m / (mx + 1e-12f);

    strong[i] = (keep && mn >= 0.3f) ? (unsigned char)1 : (unsigned char)0;
    weak[i]   = (keep && mn >= 0.1f && mn < 0.3f) ? (unsigned char)1 : (unsigned char)0;
}

// ---------------------------------------------------------------------------
// Hysteresis: block-local LDS fixpoint on a 32x32 tile with 1-px halo.
// Ping-pong global passes propagate across tile boundaries.
// ---------------------------------------------------------------------------
__global__ __launch_bounds__(256) void k_hyst(const unsigned char* __restrict__ sin_,
                                              unsigned char* __restrict__ sout,
                                              const unsigned char* __restrict__ weak)
{
    __shared__ unsigned char sS[34][34];
    __shared__ unsigned char sW[34][34];
    __shared__ int chg;

    const int tx = threadIdx.x, ty = threadIdx.y;
    const int tid = ty * 32 + tx;
    const int bx0 = blockIdx.x * 32, by0 = blockIdx.y * 32;
    const size_t ib = (size_t)blockIdx.z * NH * NW;

    for (int idx = tid; idx < 34 * 34; idx += 256) {
        int ly = idx / 34, lx = idx - ly * 34;
        int gy = by0 + ly - 1, gx = bx0 + lx - 1;
        bool in = (gy >= 0 && gy < NH && gx >= 0 && gx < NW);
        size_t g = ib + (size_t)gy * NW + gx;
        sS[ly][lx] = in ? sin_[g] : (unsigned char)0;
        sW[ly][lx] = in ? weak[g] : (unsigned char)0;
    }
    __syncthreads();

    for (int it = 0; it < 48; ++it) {
        if (tid == 0) chg = 0;
        __syncthreads();
#pragma unroll
        for (int r = 0; r < 4; ++r) {
            int ly = 1 + ty + 8 * r;
            int lx = 1 + tx;
            if (!sS[ly][lx] && sW[ly][lx]) {
                int nb = sS[ly-1][lx-1] | sS[ly-1][lx] | sS[ly-1][lx+1]
                       | sS[ly  ][lx-1]               | sS[ly  ][lx+1]
                       | sS[ly+1][lx-1] | sS[ly+1][lx] | sS[ly+1][lx+1];
                if (nb) { sS[ly][lx] = 1; chg = 1; }
            }
        }
        __syncthreads();
        if (chg == 0) break;
    }

#pragma unroll
    for (int r = 0; r < 4; ++r) {
        int ly = 1 + ty + 8 * r, lx = 1 + tx;
        sout[ib + (size_t)(by0 + ly - 1) * NW + (bx0 + lx - 1)] = sS[ly][lx];
    }
}

__global__ void k_out(const unsigned char* __restrict__ s, float* __restrict__ out)
{
    const int i = blockIdx.x * blockDim.x + threadIdx.x;
    if (i < NPIX) out[i] = (float)s[i];
}

// ---------------------------------------------------------------------------
extern "C" void kernel_launch(void* const* d_in, const int* in_sizes, int n_in,
                              void* d_out, int out_size, void* d_ws, size_t ws_size,
                              hipStream_t stream)
{
    (void)in_sizes; (void)n_in; (void)out_size; (void)ws_size;

    const float* x = (const float*)d_in[0];
    float* out = (float*)d_out;
    char* ws = (char*)d_ws;

    const size_t F  = (size_t)NPIX * sizeof(float);   // 33,554,432 B
    const size_t Bn = (size_t)NPIX;                   //  8,388,608 B

    // Workspace layout (needs ~75.5 MB), with lifetime-based reuse:
    float*         blur = (float*)(ws);               // [0, F)        live: blur..sobel
    float*         mag  = (float*)(ws + F);           // [F, 2F)       live: sobel..nms
    unsigned char* cls  = (unsigned char*)(ws + 2*F); // [2F, 2F+Bn)   live: sobel..nms
    unsigned int*  mmax = (unsigned int*)(ws + 2*F + Bn); // 128 B
    // strong/weak maps reuse the (dead) blur region:
    unsigned char* s0 = (unsigned char*)(ws);
    unsigned char* wk = (unsigned char*)(ws + Bn);
    unsigned char* s1 = (unsigned char*)(ws + 2 * Bn);

    const int nblk = NPIX / 256;

    k_init_max<<<1, 32, 0, stream>>>(mmax);
    k_blur<<<dim3(NW / 16, NH / 16, NB), 32, 0, stream>>>(x, blur);
    k_sobel<<<nblk, 256, 0, stream>>>(blur, mag, cls, mmax);
    k_nms<<<nblk, 256, 0, stream>>>(mag, cls, mmax, s0, wk);

    // 12 ping-pong passes (even count -> final result lands back in s0)
    for (int p = 0; p < 12; ++p) {
        const unsigned char* si = (p & 1) ? s1 : s0;
        unsigned char*       so = (p & 1) ? s0 : s1;
        k_hyst<<<dim3(NW / 32, NH / 32, NB), dim3(32, 8), 0, stream>>>(si, so, wk);
    }

    k_out<<<nblk, 256, 0, stream>>>(s0, out);
}
